// SSM_fusion_81716047774071
// MI455X (gfx1250) — matrix operand order repaired and checked by reading, not compiled, not run on hardware
//
#include <hip/hip_runtime.h>
#include <hip/hip_fp16.h>
#include <math.h>

typedef __attribute__((ext_vector_type(16))) _Float16 v16h;
typedef __attribute__((ext_vector_type(8)))  _Float16 v8h;
typedef __attribute__((ext_vector_type(8)))  float    v8f;

#define D_IN  32   // d_inner
#define D_ST  16   // d_state
#define KDIR  2    // scan directions
#define NB    2    // batch
#define CLEN  128  // scan chunk length

__device__ __forceinline__ v8f wmma32(v16h a, v16h b, v8f c) {
  // D = A(16x32 f16) * B(32x16 f16) + C(16x16 f32)
  return __builtin_amdgcn_wmma_f32_16x16x32_f16(false, a, false, b, (short)0, c,
                                                false, false);
}

// ================================================================ weight packs
// Conv weights OIHW f32 -> WMMA A-fragments, K order = tap-major, cin-minor.
// Layout: [mtile(2)][tap(9)*nks + ks][lane(32)][elem(16)]  (contiguous v16h/lane)
__global__ __launch_bounds__(256)
void pack_conv_frag(const float* __restrict__ w, _Float16* __restrict__ wfr,
                    int Cin) {
  int i = blockIdx.x * 256 + threadIdx.x;
  int total = 32 * Cin * 9;
  if (i >= total) return;
  const int nks  = Cin >> 5;
  int j    = i & 15;
  int lane = (i >> 4) & 31;
  int t    = i >> 9;                       // mtile*9*nks + tap*nks + ks
  int mtile = t / (9 * nks);
  int rem   = t - mtile * 9 * nks;
  int tap = rem / nks;
  int ks  = rem - tap * nks;
  int half = lane >> 4, m = lane & 15;
  int kk  = half * 16 + j; // A intra-K order
  int cin = ks * 32 + kk;
  int oc  = mtile * 16 + m;
  int ky = tap / 3, kx = tap - (tap / 3) * 3;
  wfr[i] = (_Float16)w[(((size_t)oc * Cin + cin) * 3 + ky) * 3 + kx];
}

// Generic [Mtot x 32] f32 row-major -> A-fragments [tile][lane][16]
__global__ __launch_bounds__(256)
void pack_gemm_frag(const float* __restrict__ src, _Float16* __restrict__ dst,
                    int Mtot) {
  int i = blockIdx.x * 256 + threadIdx.x;
  if (i >= Mtot * 32) return;
  int j    = i & 15;
  int lane = (i >> 4) & 31;
  int tile = i >> 9;
  int half = lane >> 4, mm = lane & 15;
  int kk = half * 16 + j;
  dst[i] = (_Float16)src[((size_t)tile * 16 + mm) * 32 + kk];
}

// ======================================================= layout / fill kernels
// zero-fill (16B per thread) -- clears halo-padded activation buffers
__global__ __launch_bounds__(256)
void zero_h8(_Float16* __restrict__ p, int n8) {
  int i = blockIdx.x * 256 + threadIdx.x;
  if (i < n8) { v8h z = {}; *(v8h*)(p + (size_t)i * 8) = z; }
}

// NCHW f32 -> halo-padded NHWC f16 at channel offset coff.
// dst layout [B][H+2][W+2][Ctot], interior at (+1,+1).
__global__ __launch_bounds__(256)
void nchw2nhwc_h(const float* __restrict__ src, _Float16* __restrict__ dst,
                 int C, int H, int W, int Ctot, int coff) {
  int i = blockIdx.x * 256 + threadIdx.x;
  int HW = H * W;
  int total = NB * C * HW;
  if (i >= total) return;
  int b = i / (C * HW);
  int r = i - b * (C * HW);
  int c = r / HW;
  int p = r - c * HW;
  int y = p / W, x = p - y * W;
  dst[(((size_t)b * (H + 2) + y + 1) * (W + 2) + x + 1) * Ctot + coff + c] =
      (_Float16)src[i];
}

// ------------------------------------- bicubic upsample -> halo-padded NHWC f16
__device__ __forceinline__ float cubw(float d) {
  const float a = -0.75f;
  float d2 = d * d, d3 = d2 * d;
  if (d <= 1.0f) return (a + 2.0f) * d3 - (a + 3.0f) * d2 + 1.0f;
  if (d < 2.0f)  return a * d3 - 5.0f * a * d2 + 8.0f * a * d - 4.0f * a;
  return 0.0f;
}

__global__ __launch_bounds__(256)
void upsample_bicubic_h(const float* __restrict__ src, _Float16* __restrict__ dst,
                        int h, int H, int W, int Ctot) {
  int idx = blockIdx.x * 256 + threadIdx.x;
  int total = NB * D_IN * H * W;
  if (idx >= total) return;
  int px = idx % W; int t = idx / W;
  int py = t % H; t /= H;
  int c = t % D_IN; int b = t / D_IN;
  float sy = (float)py * (float)(h - 1) / (float)(H - 1);
  float sx = (float)px * (float)(h - 1) / (float)(W - 1);
  int iy = (int)floorf(sy), ix = (int)floorf(sx);
  float acc = 0.0f;
#pragma unroll
  for (int dy = -1; dy <= 2; ++dy) {
    int yy = iy + dy; yy = yy < 0 ? 0 : (yy > h - 1 ? h - 1 : yy);
    float wy = cubw(fabsf(sy - (float)(iy + dy)));
#pragma unroll
    for (int dx = -1; dx <= 2; ++dx) {
      int xx = ix + dx; xx = xx < 0 ? 0 : (xx > h - 1 ? h - 1 : xx);
      float wx = cubw(fabsf(sx - (float)(ix + dx)));
      acc += wy * wx * src[((size_t)(b * D_IN + c) * h + yy) * h + xx];
    }
  }
  dst[(((size_t)b * (H + 2) + py + 1) * (W + 2) + px + 1) * Ctot + c] =
      (_Float16)acc;
}

// ===================================================== 3x3 conv, implicit WMMA
// Input: halo-padded NHWC f16 [B][H+2][W+2][Cin] (Cin mult of 32, zero halo).
// Output: NHWC f16 [B][H][W][32].
// M = 32 outCh (2 tiles), K = 9 taps x Cin, N = B*H*W. 1 wave = 16x16 tile.
// Inner loop is pure: 2x b128 A-load + 2x b128 B-load + 1 wmma (no predicates).
__global__ __launch_bounds__(128)
void conv3x3_wmma(const _Float16* __restrict__ xin, const _Float16* __restrict__ wfr,
                  const float* __restrict__ bias, _Float16* __restrict__ outp,
                  int Cin, int H, int W) {
  const int lane  = threadIdx.x & 31;
  const int wv    = threadIdx.x >> 5;
  const int mtile = wv & 1;
  const int ntile = wv >> 1;
  const int half  = lane >> 4;
  const int l16   = lane & 15;
  const int HW  = H * W;
  const int pos = blockIdx.x * 32 + ntile * 16 + l16;   // grid is exact multiple
  const int b   = pos / HW;
  const int hw  = pos - b * HW;
  const int py  = hw / W;
  const int px  = hw - py * W;
  const int nks = Cin >> 5;
  const int Wp  = W + 2;

  const _Float16* abase = wfr + ((size_t)mtile * 9 * nks * 32 + lane) * 16;

  v8f acc = {};
  for (int tap = 0; tap < 9; ++tap) {
    const int ky = tap / 3;
    const int kx = tap - (tap / 3) * 3;
    const _Float16* xrow =
        xin + (((size_t)(b * (H + 2) + py + ky)) * Wp + px + kx) * Cin + half * 16;
    const _Float16* arow = abase + (size_t)tap * nks * 32 * 16;
    for (int ks = 0; ks < nks; ++ks) {
      v16h a  = *(const v16h*)(arow + (size_t)ks * 32 * 16);  // 32B/lane coalesced
      v16h bm = *(const v16h*)(xrow + ks * 32);               // 16 contiguous cin
      acc = wmma32(a, bm, acc);
    }
  }
  const int ocb = mtile * 16 + 8 * half;
  v8h o;
#pragma unroll
  for (int v = 0; v < 8; ++v) o[v] = (_Float16)(acc[v] + bias[ocb + v]);
  *(v8h*)(outp + (size_t)pos * D_IN + ocb) = o;               // 16B store
}

// ==================== x_proj: xdbl[c,l] = Wx[c,:] . xs[:,l]  (M=64,K=32,N=L)
// dt_rank rows (0..31) -> xdt f16 [bk][l][32]; B/C rows (32..63) -> xbc f32 [bk][32][L]
__global__ __launch_bounds__(128)
void xproj_wmma(const _Float16* __restrict__ xcv, const _Float16* __restrict__ wxfr,
                _Float16* __restrict__ xdt, float* __restrict__ xbc,
                int H, int W) {
  const int lane = threadIdx.x & 31;
  const int ct   = threadIdx.x >> 5;   // c-tile 0..3
  const int half = lane >> 4;
  const int l16  = lane & 15;
  const int bk = blockIdx.y;           // b*2 + k
  const int b = bk >> 1, k = bk & 1;
  const int L = H * W;
  const int l = blockIdx.x * 16 + l16;
  const int pos = (k == 0) ? l : ((l - (l / H) * H) * W + (l / H)); // y*W+x

  v16h a = *(const v16h*)(wxfr + ((size_t)(k * 4 + ct) * 32 + lane) * 16);
  v16h bm = *(const v16h*)(xcv + ((size_t)(b * H * W) + pos) * D_IN + half * 16);
  v8f acc = {};
  acc = wmma32(a, bm, acc);

  const int cb = ct * 16 + 8 * half;
  if (ct < 2) {                       // dts part, transposed f16
    v8h o;
#pragma unroll
    for (int v = 0; v < 8; ++v) o[v] = (_Float16)acc[v];
    *(v8h*)(xdt + ((size_t)bk * L + l) * D_IN + cb) = o;
  } else {                            // B (0..15) and C (16..31) rows, f32 [n][L]
#pragma unroll
    for (int v = 0; v < 8; ++v)
      xbc[((size_t)bk * 32 + (cb - 32) + v) * L + l] = acc[v];
  }
}

// ==================== dt_proj + bias + softplus  (M=32,K=32,N=L)
__global__ __launch_bounds__(128)
void dtproj_wmma(const _Float16* __restrict__ xdt, const _Float16* __restrict__ wdtfr,
                 const float* __restrict__ dtb, float* __restrict__ delta, int L) {
  const int lane  = threadIdx.x & 31;
  const int wv    = threadIdx.x >> 5;
  const int mtile = wv & 1;
  const int ntile = wv >> 1;
  const int half  = lane >> 4;
  const int l16   = lane & 15;
  const int bk = blockIdx.y; const int k = bk & 1;
  const int l = blockIdx.x * 32 + ntile * 16 + l16;

  v16h a = *(const v16h*)(wdtfr + ((size_t)(k * 2 + mtile) * 32 + lane) * 16);
  v16h bm = *(const v16h*)(xdt + ((size_t)bk * L + l) * D_IN + half * 16);
  v8f acc = {};
  acc = wmma32(a, bm, acc);

  const int db = mtile * 16 + 8 * half;
#pragma unroll
  for (int v = 0; v < 8; ++v) {
    int d = db + v;
    float t = acc[v] + dtb[k * D_IN + d];
    float sp = (t > 20.0f) ? t : log1pf(expf(t));  // softplus
    delta[((size_t)bk * D_IN + d) * L + l] = sp;
  }
}

// ================================================== chunked selective scan
// Phase A: per chunk, local state h and sum(delta) (chunk decay = exp(A*sumDelta)).
__global__ __launch_bounds__(512)
void scan_chunk_local(const float* __restrict__ delta, const float* __restrict__ xbc,
                      const _Float16* __restrict__ xcv, const float* __restrict__ A_logs,
                      float* __restrict__ Ssum, float* __restrict__ hloc,
                      int H, int W, int NC) {
  const int tid = threadIdx.x;
  const int d = tid >> 4, n = tid & 15;
  const int c = blockIdx.x;
  const int bk = blockIdx.y; const int b = bk >> 1, k = bk & 1;
  const int L = H * W;
  const float Adn = -expf(A_logs[(k * D_IN + d) * D_ST + n]);
  const float* drow = delta + ((size_t)bk * D_IN + d) * L;
  const float* Brow = xbc + ((size_t)bk * 32 + n) * L;
  const _Float16* xb = xcv + (size_t)b * L * D_IN;
  float h = 0.0f, sdt = 0.0f;
  const int l0 = c * CLEN, l1 = l0 + CLEN;
  int py, px;
  if (k == 0) { py = l0 / W; px = l0 - py * W; }
  else        { px = l0 / H; py = l0 - px * H; }
  for (int l = l0; l < l1; ++l) {
    float dt = drow[l];
    float u  = (float)xb[((size_t)py * W + px) * D_IN + d];
    h = expf(dt * Adn) * h + dt * u * Brow[l];
    sdt += dt;
    if (k == 0) { if (++px == W) { px = 0; ++py; } }
    else        { if (++py == H) { py = 0; ++px; } }
  }
  hloc[(((size_t)bk * D_IN + d) * D_ST + n) * NC + c] = h;
  if (n == 0) Ssum[((size_t)bk * D_IN + d) * NC + c] = sdt;
}

// Phase B: short prefix across chunk summaries (2048 chains x NC steps).
__global__ __launch_bounds__(256)
void scan_chunk_prefix(const float* __restrict__ Ssum, const float* __restrict__ hloc,
                       const float* __restrict__ A_logs, float* __restrict__ Hpre,
                       int NC) {
  int idx = blockIdx.x * 256 + threadIdx.x;
  if (idx >= NB * KDIR * D_IN * D_ST) return;
  const int n = idx & 15, d = (idx >> 4) & 31, bk = idx >> 9;
  const int k = bk & 1;
  const float Adn = -expf(A_logs[(k * D_IN + d) * D_ST + n]);
  const float* Sr = Ssum + ((size_t)bk * D_IN + d) * NC;
  const float* hr = hloc + (((size_t)bk * D_IN + d) * D_ST + n) * NC;
  float*       Hr = Hpre + (((size_t)bk * D_IN + d) * D_ST + n) * NC;
  float Hc = 0.0f;
  for (int c = 0; c < NC; ++c) {
    Hr[c] = Hc;                                   // state entering chunk c
    Hc = expf(Adn * Sr[c]) * Hc + hr[c];
  }
}

// Phase C: replay chunk seeded with exact prefix state; y = C.h reduced over n.
// ys layout [bk][l][d] for coalesced LayerNorm reads.
__global__ __launch_bounds__(512)
void scan_chunk_final(const float* __restrict__ delta, const float* __restrict__ xbc,
                      const _Float16* __restrict__ xcv, const float* __restrict__ A_logs,
                      const float* __restrict__ Hpre, const float* __restrict__ Ds,
                      float* __restrict__ ys, int H, int W, int NC) {
  const int tid = threadIdx.x;
  const int d = tid >> 4, n = tid & 15;
  const int c = blockIdx.x;
  const int bk = blockIdx.y; const int b = bk >> 1, k = bk & 1;
  const int L = H * W;
  const float Adn = -expf(A_logs[(k * D_IN + d) * D_ST + n]);
  const float Dd  = Ds[k * D_IN + d];
  const float* drow = delta + ((size_t)bk * D_IN + d) * L;
  const float* Brow = xbc + ((size_t)bk * 32 + n) * L;
  const float* Crow = xbc + ((size_t)bk * 32 + 16 + n) * L;
  const _Float16* xb = xcv + (size_t)b * L * D_IN;
  float* yb = ys + (size_t)bk * L * D_IN;
  float h = Hpre[(((size_t)bk * D_IN + d) * D_ST + n) * NC + c];
  const int l0 = c * CLEN, l1 = l0 + CLEN;
  int py, px;
  if (k == 0) { py = l0 / W; px = l0 - py * W; }
  else        { px = l0 / H; py = l0 - px * H; }
  for (int l = l0; l < l1; ++l) {
    float dt = drow[l];
    float u  = (float)xb[((size_t)py * W + px) * D_IN + d];
    h = expf(dt * Adn) * h + dt * u * Brow[l];
    float yv = h * Crow[l];
    yv += __shfl_xor(yv, 1, 16);
    yv += __shfl_xor(yv, 2, 16);
    yv += __shfl_xor(yv, 4, 16);
    yv += __shfl_xor(yv, 8, 16);
    if (n == 0) yb[(size_t)l * D_IN + d] = yv + Dd * u;
    if (k == 0) { if (++px == W) { px = 0; ++py; } }
    else        { if (++py == H) { py = 0; ++px; } }
  }
}

// ===================== y0 + transpose(y1), LayerNorm (lane = channel, wave32)
__global__ __launch_bounds__(256)
void combine_ln(const float* __restrict__ ys, const float* __restrict__ g,
                const float* __restrict__ be, float* __restrict__ outp,
                int H, int W) {
  const int d = threadIdx.x;          // 0..31
  const int L = H * W;
  const int p = blockIdx.x * blockDim.y + threadIdx.y;
  if (p >= NB * L) return;            // p uniform per wave -> shuffle-safe
  const int b = p / L, l = p - b * L;
  const int py = l / W, px = l - py * W;
  float v = ys[((size_t)(b * 2 + 0) * L + l) * D_IN + d]
          + ys[((size_t)(b * 2 + 1) * L + (px * H + py)) * D_IN + d];
  float mu = v;
  mu += __shfl_xor(mu, 1, 32);  mu += __shfl_xor(mu, 2, 32);
  mu += __shfl_xor(mu, 4, 32);  mu += __shfl_xor(mu, 8, 32);
  mu += __shfl_xor(mu, 16, 32);
  mu *= (1.0f / 32.0f);
  float dv = v - mu;
  float vr = dv * dv;
  vr += __shfl_xor(vr, 1, 32);  vr += __shfl_xor(vr, 2, 32);
  vr += __shfl_xor(vr, 4, 32);  vr += __shfl_xor(vr, 8, 32);
  vr += __shfl_xor(vr, 16, 32);
  vr *= (1.0f / 32.0f);
  outp[(((size_t)b * D_IN + d) * H + py) * W + px] =
      dv * rsqrtf(vr + 1e-5f) * g[d] + be[d];
}

// NHWC f16 conv output -> NCHW f32 fusion (only used for conv0 result, 16x16)
__global__ __launch_bounds__(256)
void conv_out_to_fusion(const _Float16* __restrict__ src, float* __restrict__ dst,
                        int H) {
  int i = blockIdx.x * 256 + threadIdx.x;
  int HW = H * H;
  int total = NB * D_IN * HW;
  if (i >= total) return;
  int p = i % HW; int t = i / HW;
  int c = t % D_IN; int b = t / D_IN;
  dst[i] = (float)src[((size_t)b * HW + p) * D_IN + c];
}

// =================================================================== launcher
extern "C" void kernel_launch(void* const* d_in, const int* in_sizes, int n_in,
                              void* d_out, int out_size, void* d_ws, size_t ws_size,
                              hipStream_t stream) {
  (void)in_sizes; (void)n_in; (void)out_size; (void)ws_size;

  const float* out3    = (const float*)d_in[3];
  const float* conv0_w = (const float*)d_in[4];
  const float* conv0_b = (const float*)d_in[5];
  const float* xproj_w = (const float*)d_in[12];
  const float* dtw     = (const float*)d_in[13];
  const float* dtb     = (const float*)d_in[14];
  const float* A_logs  = (const float*)d_in[15];
  const float* Ds      = (const float*)d_in[16];
  const float* ln_g    = (const float*)d_in[17];
  const float* ln_b    = (const float*)d_in[18];

  char* ws = (char*)d_ws;
  size_t off = 0;
  auto bump = [&](size_t bytes) {
    size_t r = off; off += (bytes + 255) & ~(size_t)255; return r;
  };

  _Float16* wfr0 = (_Float16*)(ws + bump(sizeof(_Float16) * 32 * 768 * 9));
  _Float16* wfr1 = (_Float16*)(ws + bump(sizeof(_Float16) * 32 * 416 * 9));
  _Float16* wfr2 = (_Float16*)(ws + bump(sizeof(_Float16) * 32 * 224 * 9));
  _Float16* wfr3 = (_Float16*)(ws + bump(sizeof(_Float16) * 32 * 128 * 9));
  _Float16* wxfr = (_Float16*)(ws + bump(sizeof(_Float16) * KDIR * 64 * 32));
  _Float16* wdtf = (_Float16*)(ws + bump(sizeof(_Float16) * KDIR * 32 * 32));

  // halo-padded NHWC f16 activation buffers, one per conv input
  const int n0 = NB * 18 * 18 * 768;
  const int n1 = NB * 34 * 34 * 416;
  const int n2 = NB * 66 * 66 * 224;
  const int n3 = NB * 130 * 130 * 128;
  _Float16* p0 = (_Float16*)(ws + bump(sizeof(_Float16) * n0));
  _Float16* p1 = (_Float16*)(ws + bump(sizeof(_Float16) * n1));
  _Float16* p2 = (_Float16*)(ws + bump(sizeof(_Float16) * n2));
  _Float16* p3 = (_Float16*)(ws + bump(sizeof(_Float16) * n3));

  float*    fusA = (float*)(ws + bump(sizeof(float) * NB * D_IN * 64 * 64));
  float*    fusB = (float*)(ws + bump(sizeof(float) * NB * D_IN * 64 * 64));
  _Float16* xcv  = (_Float16*)(ws + bump(sizeof(_Float16) * NB * 16384 * D_IN));
  _Float16* xdt  = (_Float16*)(ws + bump(sizeof(_Float16) * NB * KDIR * 16384 * D_IN));
  float*    xbc  = (float*)(ws + bump(sizeof(float) * NB * KDIR * 32 * 16384));
  float*    delt = (float*)(ws + bump(sizeof(float) * NB * KDIR * D_IN * 16384));
  float*    ysb  = (float*)(ws + bump(sizeof(float) * NB * KDIR * 16384 * D_IN));
  float*    Ssum = (float*)(ws + bump(sizeof(float) * NB * KDIR * D_IN * 128));
  float*    hloc = (float*)(ws + bump(sizeof(float) * NB * KDIR * D_IN * D_ST * 128));
  float*    Hpre = (float*)(ws + bump(sizeof(float) * NB * KDIR * D_IN * D_ST * 128));

  // ---- zero the padded buffers (halo must be 0; interior rewritten below) ----
  auto zero = [&](_Float16* p, int n) {
    int k8 = n / 8;
    zero_h8<<<(k8 + 255) / 256, 256, 0, stream>>>(p, k8);
  };
  zero(p0, n0); zero(p1, n1); zero(p2, n2); zero(p3, n3);

  // ---- pack weights into WMMA fragment order ----
  auto packc = [&](const float* s, _Float16* dpk, int cin) {
    int n = 32 * cin * 9;
    pack_conv_frag<<<(n + 255) / 256, 256, 0, stream>>>(s, dpk, cin);
  };
  packc(conv0_w, wfr0, 768);
  packc((const float*)d_in[6],  wfr1, 416);
  packc((const float*)d_in[8],  wfr2, 224);
  packc((const float*)d_in[10], wfr3, 128);
  pack_gemm_frag<<<(128 * 32 + 255) / 256, 256, 0, stream>>>(xproj_w, wxfr, 128);
  pack_gemm_frag<<<(64 * 32 + 255) / 256, 256, 0, stream>>>(dtw, wdtf, 64);

  // ---- fusion = conv0(out3): [2,32,16,16] ----
  {
    int nconv = NB * 768 * 16 * 16;
    nchw2nhwc_h<<<(nconv + 255) / 256, 256, 0, stream>>>(out3, p0, 768, 16, 16,
                                                         768, 0);
    conv3x3_wmma<<<dim3((NB * 256) / 32), 128, 0, stream>>>(
        p0, wfr0, conv0_b, xcv, 768, 16, 16);
    conv_out_to_fusion<<<(NB * D_IN * 256 + 255) / 256, 256, 0, stream>>>(
        xcv, fusA, 16);
  }

  const float* feats[3]    = {(const float*)d_in[2], (const float*)d_in[1],
                              (const float*)d_in[0]};
  const int cfeat[3]       = {384, 192, 96};
  const int Hs[3]          = {32, 64, 128};
  _Float16* pads[3]        = {p1, p2, p3};
  const _Float16* wfrs[3]  = {wfr1, wfr2, wfr3};
  const float* cbias[3]    = {(const float*)d_in[7], (const float*)d_in[9],
                              (const float*)d_in[11]};

  float* fin = fusA;
  int hcur = 16;
  for (int i = 0; i < 3; ++i) {
    const int H = Hs[i], W = H, L = H * W;
    const int Ctot = cfeat[i] + D_IN;
    const int NC = L / CLEN;                 // 8, 32, 128 chunks
    _Float16* xcat = pads[i];

    int nup = NB * D_IN * L;
    upsample_bicubic_h<<<(nup + 255) / 256, 256, 0, stream>>>(
        fin, xcat, hcur, H, W, Ctot);
    int ncp = NB * cfeat[i] * L;
    nchw2nhwc_h<<<(ncp + 255) / 256, 256, 0, stream>>>(
        feats[i], xcat, cfeat[i], H, W, Ctot, D_IN);

    conv3x3_wmma<<<dim3((NB * L) / 32), 128, 0, stream>>>(
        xcat, wfrs[i], cbias[i], xcv, Ctot, H, W);

    xproj_wmma<<<dim3(L / 16, NB * KDIR), 128, 0, stream>>>(
        xcv, wxfr, xdt, xbc, H, W);
    dtproj_wmma<<<dim3(L / 32, NB * KDIR), 128, 0, stream>>>(
        xdt, wdtf, dtb, delt, L);

    scan_chunk_local<<<dim3(NC, NB * KDIR), 512, 0, stream>>>(
        delt, xbc, xcv, A_logs, Ssum, hloc, H, W, NC);
    scan_chunk_prefix<<<8, 256, 0, stream>>>(Ssum, hloc, A_logs, Hpre, NC);
    scan_chunk_final<<<dim3(NC, NB * KDIR), 512, 0, stream>>>(
        delt, xbc, xcv, A_logs, Hpre, Ds, ysb, H, W, NC);

    float* fout = (i == 2) ? (float*)d_out : ((fin == fusA) ? fusB : fusA);
    combine_ln<<<dim3((NB * L + 7) / 8), dim3(32, 8), 0, stream>>>(
        ysb, ln_g, ln_b, fout, H, W);
    fin = fout; hcur = H;
  }
}
